// MPNN_26740466385663
// MI455X (gfx1250) — matrix operand order repaired and checked by reading, not compiled, not run on hardware
//
#include <hip/hip_runtime.h>
#include <hip/hip_bf16.h>

#define N_NODES 40000
#define N_EDGES 120000
#define NIN 74
#define D 64
#define HE 128
#define STEPS 3

typedef __bf16 v16bf __attribute__((ext_vector_type(16)));
typedef float  v8f   __attribute__((ext_vector_type(8)));
typedef __bf16 bfx8  __attribute__((ext_vector_type(8)));
typedef float  fx4   __attribute__((ext_vector_type(4)));

__device__ __forceinline__ v8f wmma_bf16(v16bf a, v16bf b, v8f c) {
  // D = A(16x32 bf16) * B(32x16 bf16) + C(16x16 f32)
  return __builtin_amdgcn_wmma_f32_16x16x32_bf16(
      false, a, false, b, (short)0, c, false, false);
}

// A fragment (16x32 bf16) from LDS, row-major [m][k] with padded stride.
// Lane layout: lanes 0-15 hold k = kb..kb+15 for row m=lane;
// lanes 16-31 hold k = kb+16..kb+31 for row m=lane-16.
__device__ __forceinline__ v16bf ldsA_bf16(const __bf16* base, int stride,
                                           int kbase, int lane) {
  int m = lane & 15;
  int ko = (lane >> 4) << 4; // 0 or 16
  const __bf16* p = base + m * stride + kbase + ko;
  bfx8 lo = *(const bfx8*)p;
  bfx8 hi = *(const bfx8*)(p + 8);
  v16bf r;
#pragma unroll
  for (int j = 0; j < 8; j++) { r[j] = lo[j]; r[j + 8] = hi[j]; }
  return r;
}

// A fragment from f32 LDS (convert to bf16 on the fly).
__device__ __forceinline__ v16bf ldsA_f32(const float* base, int stride,
                                          int kbase, int lane) {
  int m = lane & 15;
  int ko = (lane >> 4) << 4;
  const float* p = base + m * stride + kbase + ko;
  fx4 a0 = *(const fx4*)p;
  fx4 a1 = *(const fx4*)(p + 4);
  fx4 b0 = *(const fx4*)(p + 8);
  fx4 b1 = *(const fx4*)(p + 12);
  v16bf r;
#pragma unroll
  for (int j = 0; j < 4; j++) {
    r[j]      = (__bf16)a0[j];
    r[j + 4]  = (__bf16)a1[j];
    r[j + 8]  = (__bf16)b0[j];
    r[j + 12] = (__bf16)b1[j];
  }
  return r;
}

// B fragment (32x16 bf16) from a global bf16 matrix stored [outCol][k] row-major.
// Lane n holds k = kbase + (lane>=16 ? 16 : 0) + j, j=0..15 (contiguous 32B).
__device__ __forceinline__ v16bf gB_bf16(const __bf16* base, int ld, int col0,
                                         int kbase, int lane) {
  int n = lane & 15;
  int kh = (lane >> 4) << 4;
  const __bf16* p = base + (size_t)(col0 + n) * ld + kbase + kh;
  bfx8 lo = *(const bfx8*)p;
  bfx8 hi = *(const bfx8*)(p + 8);
  v16bf r;
#pragma unroll
  for (int j = 0; j < 8; j++) { r[j] = lo[j]; r[j + 8] = hi[j]; }
  return r;
}

// B fragment from a global f32 matrix stored [k][ncols] (gather down column).
__device__ __forceinline__ v16bf gB_f32_kmajor(const float* W, int ldn, int col0,
                                               int kbase, int lane) {
  int n = lane & 15;
  int kh = (lane >> 4) << 4;
  const float* p = W + (size_t)(kbase + kh) * ldn + col0 + n;
  v16bf r;
#pragma unroll
  for (int j = 0; j < 16; j++) r[j] = (__bf16)p[(size_t)j * ldn];
  return r;
}

// ---------------------------------------------------------------------------
// Kernel 1: h = relu(H @ W_proj + b_proj)   (N x 74) @ (74 x 64)
__global__ __launch_bounds__(256) void k_proj(const float* __restrict__ H,
                                              const float* __restrict__ Wp,
                                              const float* __restrict__ bp,
                                              float* __restrict__ h) {
  int idx = blockIdx.x * 256 + threadIdx.x; // over N*D (exact)
  int v = idx >> 6, o = idx & 63;
  float acc = bp[o];
  const float* hr = H + (size_t)v * NIN;
#pragma unroll 2
  for (int i = 0; i < NIN; i++) acc = fmaf(hr[i], Wp[i * D + o], acc);
  h[idx] = fmaxf(acc, 0.f);
}

// ---------------------------------------------------------------------------
// Kernel 1b: GRU weights f32 -> bf16 (kept row-major [gate][j])
__global__ __launch_bounds__(256) void k_cvt(const float* __restrict__ a,
                                             const float* __restrict__ b,
                                             __bf16* __restrict__ oa,
                                             __bf16* __restrict__ ob) {
  int i = blockIdx.x * 256 + threadIdx.x; // 12288 each (exact grid)
  oa[i] = (__bf16)a[i];
  ob[i] = (__bf16)b[i];
}

// ---------------------------------------------------------------------------
// Kernel 2: We[e][i*64+o] = relu(e*We1+be1) @ We2 + be2, stored bf16.
// Grid (125, 8): x = 960-edge chunk, y = 512-col chunk. 256 threads = 8 waves,
// each wave owns 64 columns; B fragments (K=128) hoisted into VGPRs per block.
#define EDGE_BLK 960
#define A_STR 136  // padded bf16 row stride (272B) -> conflict-free ds_b128

__global__ __launch_bounds__(256) void k_edgenet(const float* __restrict__ ev,
                                                 const float* __restrict__ We1,
                                                 const float* __restrict__ be1,
                                                 const float* __restrict__ We2,
                                                 const float* __restrict__ be2,
                                                 __bf16* __restrict__ We) {
  __shared__ __bf16 sA[16 * A_STR];
  int tid = threadIdx.x;
  int lane = tid & 31, wv = tid >> 5;
  int ebase = blockIdx.x * EDGE_BLK;
  int col0 = blockIdx.y * 512 + wv * 64;

  v16bf Bf[4][4]; // [ntile][kchunk]
#pragma unroll
  for (int nt = 0; nt < 4; nt++)
#pragma unroll
    for (int kc = 0; kc < 4; kc++)
      Bf[nt][kc] = gB_f32_kmajor(We2, 4096, col0 + nt * 16, kc * 32, lane);

  float bias[4];
#pragma unroll
  for (int nt = 0; nt < 4; nt++) bias[nt] = be2[col0 + nt * 16 + (lane & 15)];

  for (int t = 0; t < EDGE_BLK / 16; t++) {
    int e0 = ebase + t * 16;
    { // build a-tile: a[m][k] = relu(e*We1[k]+be1[k]), 8 values per thread
      int m = tid & 15, k0 = (tid >> 4) * 8;
      float e = ev[e0 + m];
#pragma unroll
      for (int j = 0; j < 8; j++) {
        int k = k0 + j;
        sA[m * A_STR + k] = (__bf16)fmaxf(fmaf(e, We1[k], be1[k]), 0.f);
      }
    }
    __syncthreads();

    v8f C[4];
#pragma unroll
    for (int nt = 0; nt < 4; nt++) {
      v8f z;
#pragma unroll
      for (int v = 0; v < 8; v++) z[v] = bias[nt];
      C[nt] = z;
    }
#pragma unroll
    for (int kc = 0; kc < 4; kc++) {
      v16bf A = ldsA_bf16(sA, A_STR, kc * 32, lane);
#pragma unroll
      for (int nt = 0; nt < 4; nt++) C[nt] = wmma_bf16(A, Bf[nt][kc], C[nt]);
    }
    // store C -> bf16 We[e][col]; C layout: m = v + 8*(lane>>4), n = lane&15
    int n = lane & 15, hi = lane >> 4;
#pragma unroll
    for (int nt = 0; nt < 4; nt++)
#pragma unroll
      for (int v = 0; v < 8; v++) {
        int m = v + 8 * hi;
        We[(size_t)(e0 + m) * 4096 + col0 + nt * 16 + n] = (__bf16)C[nt][v];
      }
    __syncthreads();
  }
}

// ---------------------------------------------------------------------------
// Kernel 3: per-edge message GEMV + scatter-add.  One wave per edge.
// m[o] = sum_i h[src][i] * We[e][i][o];  atomic add into agg[dst][o].
__global__ __launch_bounds__(256) void k_message(const float* __restrict__ h,
                                                 const __bf16* __restrict__ We,
                                                 const int* __restrict__ src,
                                                 const int* __restrict__ dst,
                                                 float* __restrict__ agg) {
  __shared__ float sh[8][64];
  int lane = threadIdx.x & 31, wv = threadIdx.x >> 5;
  int eid = blockIdx.x * 8 + wv;
  int s = src[eid], d = dst[eid];

  const float* hr = h + (size_t)s * 64;
  float2 hv = *(const float2*)(hr + 2 * lane);
  sh[wv][2 * lane]     = hv.x;
  sh[wv][2 * lane + 1] = hv.y;

  float a0 = 0.f, a1 = 0.f;
  const __bf16* W = We + (size_t)eid * 4096 + 2 * lane; // lane owns cols 2L,2L+1
#pragma unroll 8
  for (int i = 0; i < 64; i++) {
    unsigned wb = *(const unsigned*)(W + i * 64); // two bf16, coalesced 128B/row
    float w0 = __builtin_bit_cast(float, wb << 16);
    float w1 = __builtin_bit_cast(float, wb & 0xffff0000u);
    float hi = sh[wv][i]; // LDS broadcast
    a0 = fmaf(hi, w0, a0);
    a1 = fmaf(hi, w1, a1);
  }
  float* ap = agg + (size_t)d * 64 + 2 * lane;
  unsafeAtomicAdd(ap, a0);
  unsafeAtomicAdd(ap + 1, a1);
}

// ---------------------------------------------------------------------------
// Kernel 3a: agg = conv_b (so segment-sum lands on the bias)
__global__ __launch_bounds__(256) void k_agg_init(float* __restrict__ agg,
                                                  const float* __restrict__ cb) {
  int idx = blockIdx.x * 256 + threadIdx.x;
  agg[idx] = cb[idx & 63];
}

// ---------------------------------------------------------------------------
// Kernel 4: fused GRU cell.  x = relu(agg); gi = x@Wih^T; gh = hp@Whh^T;
// r=sig, z=sig, n=tanh(gi_n + r*gh_n); out = (1-z)*n + z*hp.
// 128 threads = 4 waves, 64 nodes/block (exact: 625 blocks), wave = one M-tile.
#define XS 68  // padded f32 row stride (conflict-free ds_b128)
__device__ __forceinline__ float sigm(float x) { return 1.f / (1.f + __expf(-x)); }

__global__ __launch_bounds__(128) void k_gru(const float* __restrict__ agg,
                                             const float* __restrict__ h,
                                             const __bf16* __restrict__ Wih,
                                             const __bf16* __restrict__ Whh,
                                             const float* __restrict__ bih,
                                             const float* __restrict__ bhh,
                                             float* __restrict__ out) {
  __shared__ float sX[64 * XS];
  __shared__ float sH[64 * XS];
  int tid = threadIdx.x, lane = tid & 31, wv = tid >> 5;
  int node0 = blockIdx.x * 64;

  for (int idx = tid; idx < 64 * 64; idx += 128) {
    int r = idx >> 6, c = idx & 63;
    size_t g = (size_t)(node0 + r) * 64 + c;
    sX[r * XS + c] = fmaxf(agg[g], 0.f);
    sH[r * XS + c] = h[g];
  }
  __syncthreads();

  int m0 = wv * 16;
  int n = lane & 15, hi8 = (lane >> 4) * 8;

  float br[4], bz[4], bgi[4], bgh[4];
#pragma unroll
  for (int nt = 0; nt < 4; nt++) {
    int g = 16 * nt + n;
    br[nt]  = bih[g] + bhh[g];
    bz[nt]  = bih[64 + g] + bhh[64 + g];
    bgi[nt] = bih[128 + g];
    bgh[nt] = bhh[128 + g];
  }

  v16bf Ax[2], Ah[2];
#pragma unroll
  for (int kc = 0; kc < 2; kc++) {
    Ax[kc] = ldsA_f32(sX + m0 * XS, XS, kc * 32, lane);
    Ah[kc] = ldsA_f32(sH + m0 * XS, XS, kc * 32, lane);
  }

  v8f R[4], Z[4], GI[4], GH[4];
#pragma unroll
  for (int nt = 0; nt < 4; nt++) {
    v8f a, b, c, d;
#pragma unroll
    for (int v = 0; v < 8; v++) { a[v]=br[nt]; b[v]=bz[nt]; c[v]=bgi[nt]; d[v]=bgh[nt]; }
    R[nt]=a; Z[nt]=b; GI[nt]=c; GH[nt]=d;
  }

#pragma unroll
  for (int kc = 0; kc < 2; kc++) {
    int kb = kc * 32;
#pragma unroll
    for (int nt = 0; nt < 4; nt++) {
      int g0 = 16 * nt;
      R[nt]  = wmma_bf16(Ax[kc], gB_bf16(Wih, 64, g0,        kb, lane), R[nt]);
      R[nt]  = wmma_bf16(Ah[kc], gB_bf16(Whh, 64, g0,        kb, lane), R[nt]);
      Z[nt]  = wmma_bf16(Ax[kc], gB_bf16(Wih, 64, 64 + g0,   kb, lane), Z[nt]);
      Z[nt]  = wmma_bf16(Ah[kc], gB_bf16(Whh, 64, 64 + g0,   kb, lane), Z[nt]);
      GI[nt] = wmma_bf16(Ax[kc], gB_bf16(Wih, 64, 128 + g0,  kb, lane), GI[nt]);
      GH[nt] = wmma_bf16(Ah[kc], gB_bf16(Whh, 64, 128 + g0,  kb, lane), GH[nt]);
    }
  }

#pragma unroll
  for (int nt = 0; nt < 4; nt++)
#pragma unroll
    for (int v = 0; v < 8; v++) {
      int m = m0 + v + hi8;
      float r_ = sigm(R[nt][v]);
      float z_ = sigm(Z[nt][v]);
      float n_ = tanhf(GI[nt][v] + r_ * GH[nt][v]);
      float hp = sH[m * XS + 16 * nt + n];
      out[(size_t)(node0 + m) * 64 + 16 * nt + n] = (1.f - z_) * n_ + z_ * hp;
    }
}

// ---------------------------------------------------------------------------
extern "C" void kernel_launch(void* const* d_in, const int* in_sizes, int n_in,
                              void* d_out, int out_size, void* d_ws, size_t ws_size,
                              hipStream_t stream) {
  const float* H     = (const float*)d_in[0];
  const float* ev    = (const float*)d_in[1];
  const int*   src   = (const int*)d_in[2];
  const int*   dst   = (const int*)d_in[3];
  const float* Wp    = (const float*)d_in[4];
  const float* bp    = (const float*)d_in[5];
  const float* We1   = (const float*)d_in[6];
  const float* be1   = (const float*)d_in[7];
  const float* We2   = (const float*)d_in[8];
  const float* be2   = (const float*)d_in[9];
  const float* convb = (const float*)d_in[10];
  const float* Wih   = (const float*)d_in[11];
  const float* Whh   = (const float*)d_in[12];
  const float* bih   = (const float*)d_in[13];
  const float* bhh   = (const float*)d_in[14];

  char* ws = (char*)d_ws;
  auto carve = [&](size_t bytes) {
    char* p = ws;
    ws += (bytes + 255) & ~(size_t)255;
    return p;
  };
  float*  h    = (float*)carve((size_t)N_NODES * D * sizeof(float));
  float*  agg  = (float*)carve((size_t)N_NODES * D * sizeof(float));
  __bf16* WihB = (__bf16*)carve((size_t)3 * D * D * sizeof(__bf16));
  __bf16* WhhB = (__bf16*)carve((size_t)3 * D * D * sizeof(__bf16));
  __bf16* WeB  = (__bf16*)carve((size_t)N_EDGES * D * D * sizeof(__bf16));

  k_proj<<<(N_NODES * D) / 256, 256, 0, stream>>>(H, Wp, bp, h);
  k_cvt<<<(3 * D * D) / 256, 256, 0, stream>>>(Wih, Whh, WihB, WhhB);

  dim3 g2(N_EDGES / EDGE_BLK, (D * D) / 512); // (125, 8)
  k_edgenet<<<g2, 256, 0, stream>>>(ev, We1, be1, We2, be2, WeB);

  for (int s = 0; s < STEPS; s++) {
    k_agg_init<<<(N_NODES * D) / 256, 256, 0, stream>>>(agg, convb);
    k_message<<<N_EDGES / 8, 256, 0, stream>>>(h, WeB, src, dst, agg);
    float* dstp = (s == STEPS - 1) ? (float*)d_out : h;
    k_gru<<<N_NODES / 64, 128, 0, stream>>>(agg, h, WihB, WhhB, bih, bhh, dstp);
  }
}
